// VGAE_31903017074975
// MI455X (gfx1250) — matrix operand order repaired and checked by reading, not compiled, not run on hardware
//
#include <hip/hip_runtime.h>
#include <hip/hip_bf16.h>

#define NPTS 10000
#define DIN  512
#define H1   256
#define H2   16
#define KCL  10
#define MT   625          // NPTS / 16 m-tiles

typedef __attribute__((ext_vector_type(16))) __bf16 v16bf;
typedef __attribute__((ext_vector_type(8)))  float  v8f;
typedef __attribute__((ext_vector_type(4)))  float  v4f;
typedef __attribute__((ext_vector_type(2)))  float  v2f;

// ---------- workspace layout (bytes, 256-aligned) ----------
#define OFF_WT   0                         // 512*256 bf16      = 262144
#define OFF_XWT  (OFF_WT  + 262144)        // 256*10000 bf16    = 5120000; 5000192 reserved
#define OFF_HID  (OFF_XWT + 5000192)       // 10000*256 f32     = 10240000
#define OFF_P1T  (OFF_HID + 10240000)      // 16*10000 bf16     = 320000
#define OFF_PI   (OFF_P1T + 320000)        // 10000*16 f32      = 640000
#define OFF_Z    (OFF_PI  + 640000)        // 10000*16 f32      = 640000
#define OFF_SQ   (OFF_Z   + 640000)        // 10000 f32         = 40000

#define WMMA_BF16(A, B, C) \
  __builtin_amdgcn_wmma_f32_16x16x32_bf16(false, (A), false, (B), (short)0, (C), false, false)

// ---------- small helpers ----------
__device__ __forceinline__ v8f f_zero8() {
  v8f v;
#pragma unroll
  for (int i = 0; i < 8; ++i) v[i] = 0.0f;
  return v;
}
__device__ __forceinline__ v16bf bf_zero16() {
  v16bf v;
#pragma unroll
  for (int i = 0; i < 16; ++i) v[i] = (__bf16)0.0f;
  return v;
}
__device__ __forceinline__ v4f ld_nt4(const float* p) {
  return __builtin_nontemporal_load((const v4f*)p);
}
__device__ __forceinline__ v16bf cvt_afrag(v4f a0, v4f a1, v4f b0, v4f b1) {
  v16bf v;
  v[0]=(__bf16)a0.x;  v[1]=(__bf16)a0.y;  v[2]=(__bf16)a0.z;  v[3]=(__bf16)a0.w;
  v[4]=(__bf16)a1.x;  v[5]=(__bf16)a1.y;  v[6]=(__bf16)a1.z;  v[7]=(__bf16)a1.w;
  v[8]=(__bf16)b0.x;  v[9]=(__bf16)b0.y;  v[10]=(__bf16)b0.z; v[11]=(__bf16)b0.w;
  v[12]=(__bf16)b1.x; v[13]=(__bf16)b1.y; v[14]=(__bf16)b1.z; v[15]=(__bf16)b1.w;
  return v;
}
// A fragment 16x32 bf16 from a streamed (non-temporal) f32 row: lane(hf,r) = row m0+r,
// elems 0-15 -> K = k0+hf*16+{0..15}
__device__ __forceinline__ v16bf load_afrag_nt(const float* __restrict__ row, int k0, int hf) {
  const float* p1 = row + k0 + hf * 16;
  const float* p2 = p1 + 8;
  return cvt_afrag(ld_nt4(p1), ld_nt4(p1 + 4), ld_nt4(p2), ld_nt4(p2 + 4));
}
// zero-padded tail A fragment: only hf==0 lanes, K = k0+{0..15} valid (K tail of 16)
__device__ __forceinline__ v16bf load_afrag_nt_tail(const float* __restrict__ row, int k0, int hf) {
  const float* p1 = row + k0;
  v4f z = {0.0f, 0.0f, 0.0f, 0.0f};
  if (hf) return cvt_afrag(z, z, z, z);
  return cvt_afrag(ld_nt4(p1), ld_nt4(p1 + 4), ld_nt4(p1 + 8), ld_nt4(p1 + 12));
}
// B fragment 32x16 bf16 from K-major (transposed) storage: lane(hf,r) holds
// column n=r, K = k0+hf*16+{0..15}, contiguous 32B. Temporal (L2-resident operand).
__device__ __forceinline__ v16bf load_bfrag(const __bf16* __restrict__ p) {
  union U { uint4 u[2]; v16bf v; } x;
  const uint4* q = (const uint4*)p;
  x.u[0] = q[0];
  x.u[1] = q[1];
  return x.v;
}

// ---------- kernel 0: W_base (512x256 f32) -> Wt (256x512 bf16, K-major) ----------
__global__ void wt_convert_kernel(const float* __restrict__ W, __bf16* __restrict__ Wt) {
  int idx = blockIdx.x * 256 + threadIdx.x;           // 0 .. 512*256-1
  if (idx >= DIN * H1) return;
  int k = idx / H1, n = idx % H1;                     // W[k][n]
  Wt[(size_t)n * DIN + k] = (__bf16)W[idx];
}

// ---------- kernel 1: XW = X @ W_base, stored transposed bf16 [H1][NPTS] ----------
// 8 waves/block, 2 n-tiles (32 cols) per wave; ping-pong double buffer, K=512.
__global__ __launch_bounds__(256, 1) void gemm1_xw_kernel(
    const float* __restrict__ X, const __bf16* __restrict__ Wt, __bf16* __restrict__ XWt) {
  const int wave = threadIdx.x >> 5;                  // 0..7
  const int lane = threadIdx.x & 31;
  const int hf = lane >> 4, r = lane & 15;
  const int m0 = blockIdx.x * 16, n0 = wave * 32;

  v8f acc0 = f_zero8(), acc1 = f_zero8();
  const float* arow = X + (size_t)(m0 + r) * DIN;
  const __bf16* bb0 = Wt + (size_t)(n0 + r) * DIN;
  const __bf16* bb1 = Wt + (size_t)(n0 + 16 + r) * DIN;

  v16bf a0 = load_afrag_nt(arow, 0, hf);
  v16bf b00 = load_bfrag(bb0 + hf * 16);
  v16bf b01 = load_bfrag(bb1 + hf * 16);

  for (int k0 = 0; k0 < DIN - 64; k0 += 64) {         // pairs; preload next pair
    v16bf a1  = load_afrag_nt(arow, k0 + 32, hf);
    v16bf b10 = load_bfrag(bb0 + k0 + 32 + hf * 16);
    v16bf b11 = load_bfrag(bb1 + k0 + 32 + hf * 16);
    acc0 = WMMA_BF16(a0, b00, acc0);
    acc1 = WMMA_BF16(a0, b01, acc1);
    a0  = load_afrag_nt(arow, k0 + 64, hf);
    b00 = load_bfrag(bb0 + k0 + 64 + hf * 16);
    b01 = load_bfrag(bb1 + k0 + 64 + hf * 16);
    acc0 = WMMA_BF16(a1, b10, acc0);
    acc1 = WMMA_BF16(a1, b11, acc1);
  }
  { // peeled final pair: k = 448 (in buf0) and 480
    v16bf a1  = load_afrag_nt(arow, DIN - 32, hf);
    v16bf b10 = load_bfrag(bb0 + DIN - 32 + hf * 16);
    v16bf b11 = load_bfrag(bb1 + DIN - 32 + hf * 16);
    acc0 = WMMA_BF16(a0, b00, acc0);
    acc1 = WMMA_BF16(a0, b01, acc1);
    acc0 = WMMA_BF16(a1, b10, acc0);
    acc1 = WMMA_BF16(a1, b11, acc1);
  }
  // D layout: elem i -> (M = i + hf*8, N = r). Transposed bf16 store: 8 contiguous m.
#pragma unroll
  for (int t = 0; t < 2; ++t) {
    const v8f acc = t ? acc1 : acc0;
    const int n = n0 + t * 16 + r;
    __bf16 tmp[8];
#pragma unroll
    for (int i = 0; i < 8; ++i) tmp[i] = (__bf16)acc[i];
    *(uint4*)(XWt + (size_t)n * NPTS + m0 + hf * 8) = *(const uint4*)tmp;
  }
}

// ---------- kernel 2: hidden = relu(adj @ XW)  [NPTS][H1] f32 ----------
// 8 waves/block, 2 n-tiles/wave. K = 10000: 155 pairs (0..9888) + 9920 + 9952 + tail 9984.
__global__ __launch_bounds__(256, 1) void gemm2_hidden_kernel(
    const float* __restrict__ adj, const __bf16* __restrict__ XWt, float* __restrict__ hidden) {
  const int wave = threadIdx.x >> 5;
  const int lane = threadIdx.x & 31;
  const int hf = lane >> 4, r = lane & 15;
  const int m0 = blockIdx.x * 16, n0 = wave * 32;

  v8f acc0 = f_zero8(), acc1 = f_zero8();
  const float* arow = adj + (size_t)(m0 + r) * NPTS;
  const __bf16* bb0 = XWt + (size_t)(n0 + r) * NPTS;
  const __bf16* bb1 = XWt + (size_t)(n0 + 16 + r) * NPTS;

  v16bf a0 = load_afrag_nt(arow, 0, hf);
  v16bf b00 = load_bfrag(bb0 + hf * 16);
  v16bf b01 = load_bfrag(bb1 + hf * 16);

  for (int k0 = 0; k0 < 9920; k0 += 64) {             // k0 = 0,64,...,9856 (155 iters)
    v16bf a1  = load_afrag_nt(arow, k0 + 32, hf);
    v16bf b10 = load_bfrag(bb0 + k0 + 32 + hf * 16);
    v16bf b11 = load_bfrag(bb1 + k0 + 32 + hf * 16);
    acc0 = WMMA_BF16(a0, b00, acc0);
    acc1 = WMMA_BF16(a0, b01, acc1);
    a0  = load_afrag_nt(arow, k0 + 64, hf);
    b00 = load_bfrag(bb0 + k0 + 64 + hf * 16);
    b01 = load_bfrag(bb1 + k0 + 64 + hf * 16);
    acc0 = WMMA_BF16(a1, b10, acc0);
    acc1 = WMMA_BF16(a1, b11, acc1);
  }
  { // compute 9920 (buf0) while loading 9952; then tail 9984 (K=16, zero-padded)
    v16bf a1  = load_afrag_nt(arow, 9952, hf);
    v16bf b10 = load_bfrag(bb0 + 9952 + hf * 16);
    v16bf b11 = load_bfrag(bb1 + 9952 + hf * 16);
    acc0 = WMMA_BF16(a0, b00, acc0);
    acc1 = WMMA_BF16(a0, b01, acc1);
    v16bf at  = load_afrag_nt_tail(arow, 9984, hf);
    v16bf bt0 = bf_zero16();
    v16bf bt1 = bf_zero16();
    if (!hf) {                                        // hf==1 lanes hold K 16..31 -> zero
      bt0 = load_bfrag(bb0 + 9984);
      bt1 = load_bfrag(bb1 + 9984);
    }
    acc0 = WMMA_BF16(a1, b10, acc0);
    acc1 = WMMA_BF16(a1, b11, acc1);
    acc0 = WMMA_BF16(at, bt0, acc0);
    acc1 = WMMA_BF16(at, bt1, acc1);
  }
#pragma unroll
  for (int t = 0; t < 2; ++t) {
    const v8f acc = t ? acc1 : acc0;
    const int n = n0 + t * 16 + r;
#pragma unroll
    for (int i = 0; i < 8; ++i) {
      const int m = m0 + i + hf * 8;
      hidden[(size_t)m * H1 + n] = fmaxf(acc[i], 0.0f);
    }
  }
}

// ---------- kernel 3: P1t[j][i] = (hidden @ W_pi)[i][j], bf16, padded to 16 rows ----------
__global__ void p1_kernel(const float* __restrict__ hidden, const float* __restrict__ Wpi,
                          __bf16* __restrict__ P1t) {
  int i = blockIdx.x * 256 + threadIdx.x;
  if (i >= NPTS) return;
  const float* hrow = hidden + (size_t)i * H1;
  float s[KCL];
#pragma unroll
  for (int j = 0; j < KCL; ++j) s[j] = 0.0f;
  for (int k = 0; k < H1; ++k) {
    float h = hrow[k];
#pragma unroll
    for (int j = 0; j < KCL; ++j) s[j] = fmaf(h, Wpi[k * KCL + j], s[j]);
  }
#pragma unroll
  for (int j = 0; j < 16; ++j)
    P1t[(size_t)j * NPTS + i] = (j < KCL) ? (__bf16)s[j] : (__bf16)0.0f;
}

// ---------- kernel 4: pi = adj @ P1  [NPTS][16] f32 (cols 10..15 are zero) ----------
__global__ __launch_bounds__(128, 1) void gemm3_pi_kernel(
    const float* __restrict__ adj, const __bf16* __restrict__ P1t, float* __restrict__ pi) {
  const int wave = threadIdx.x >> 5;
  const int lane = threadIdx.x & 31;
  const int hf = lane >> 4, r = lane & 15;
  const int mt = blockIdx.x * 4 + wave;
  if (mt >= MT) return;
  const int m0 = mt * 16;

  v8f acc = f_zero8();
  const float* arow = adj + (size_t)(m0 + r) * NPTS;
  const __bf16* bbase = P1t + (size_t)r * NPTS;

  v16bf a0 = load_afrag_nt(arow, 0, hf);
  v16bf b0 = load_bfrag(bbase + hf * 16);

  for (int k0 = 0; k0 < 9920; k0 += 64) {
    v16bf a1 = load_afrag_nt(arow, k0 + 32, hf);
    v16bf b1 = load_bfrag(bbase + k0 + 32 + hf * 16);
    acc = WMMA_BF16(a0, b0, acc);
    a0 = load_afrag_nt(arow, k0 + 64, hf);
    b0 = load_bfrag(bbase + k0 + 64 + hf * 16);
    acc = WMMA_BF16(a1, b1, acc);
  }
  {
    v16bf a1 = load_afrag_nt(arow, 9952, hf);
    v16bf b1 = load_bfrag(bbase + 9952 + hf * 16);
    acc = WMMA_BF16(a0, b0, acc);
    v16bf at = load_afrag_nt_tail(arow, 9984, hf);
    v16bf bt = bf_zero16();
    if (!hf) bt = load_bfrag(bbase + 9984);
    acc = WMMA_BF16(a1, b1, acc);
    acc = WMMA_BF16(at, bt, acc);
  }
#pragma unroll
  for (int i = 0; i < 8; ++i)
    pi[(size_t)(m0 + i + hf * 8) * 16 + r] = acc[i];
}

// ---------- kernel 5: G, softmax(pi), Z = softmax @ G, sq = ||z||^2 ----------
__global__ void z_kernel(const float* __restrict__ pi, const float* __restrict__ mean,
                         const float* __restrict__ logstd, const float* __restrict__ noise,
                         float* __restrict__ Z, float* __restrict__ sq) {
  __shared__ float G[KCL * H2];
  const int t = threadIdx.x;
  if (t < KCL * H2) G[t] = mean[t] + noise[t] * __expf(logstd[t]);
  __syncthreads();
  const int i = blockIdx.x * 256 + t;
  if (i >= NPTS) return;
  float p[KCL];
  float mx = -3.402823466e38f;
#pragma unroll
  for (int j = 0; j < KCL; ++j) { p[j] = pi[(size_t)i * 16 + j]; mx = fmaxf(mx, p[j]); }
  float sum = 0.0f;
#pragma unroll
  for (int j = 0; j < KCL; ++j) { p[j] = __expf(p[j] - mx); sum += p[j]; }
  const float inv = 1.0f / sum;
  float zz[H2];
#pragma unroll
  for (int h = 0; h < H2; ++h) zz[h] = 0.0f;
#pragma unroll
  for (int j = 0; j < KCL; ++j) {
    const float w = p[j] * inv;
#pragma unroll
    for (int h = 0; h < H2; ++h) zz[h] = fmaf(w, G[j * H2 + h], zz[h]);
  }
  float s = 0.0f;
#pragma unroll
  for (int h = 0; h < H2; ++h) { Z[(size_t)i * H2 + h] = zz[h]; s = fmaf(zz[h], zz[h], s); }
  sq[i] = s;
}

// ---------- kernel 6: A_pred = sigmoid(2*Z@Z^T - sq_i - sq_j + alpha), f32 WMMA ----------
__global__ __launch_bounds__(256, 1) void apred_kernel(
    const float* __restrict__ Z, const float* __restrict__ sq,
    const float* __restrict__ alphap, float* __restrict__ out) {
  const int wave = threadIdx.x >> 5;
  const int lane = threadIdx.x & 31;
  const int hf = lane >> 4, r = lane & 15;
  const int it = blockIdx.y * 8 + wave;               // i-tile
  if (it >= MT) return;
  const int i0 = it * 16;
  const int jg = blockIdx.x;                          // group of 4 j-tiles
  const float alpha = *alphap;

  const v4f* zi = (const v4f*)(Z + (size_t)(i0 + r) * H2);
  v4f ziv[4];
#pragma unroll
  for (int q = 0; q < 4; ++q) ziv[q] = zi[q];
  float sqi[8];
#pragma unroll
  for (int i = 0; i < 8; ++i) sqi[i] = sq[i0 + i + hf * 8];

#pragma unroll
  for (int t = 0; t < 4; ++t) {
    const int jt = jg * 4 + t;
    if (jt >= MT) continue;                           // wave-uniform
    const int j0 = jt * 16;
    const v4f* zj = (const v4f*)(Z + (size_t)(j0 + r) * H2);
    v4f zjv[4];
#pragma unroll
    for (int q = 0; q < 4; ++q) zjv[q] = zj[q];

    v8f acc = f_zero8();
#pragma unroll
    for (int c = 0; c < 4; ++c) {                     // K = c*4 + hf*2 + {0,1}
      v2f a, b;
      a[0] = hf ? ziv[c].z : ziv[c].x;
      a[1] = hf ? ziv[c].w : ziv[c].y;
      b[0] = hf ? zjv[c].z : zjv[c].x;
      b[1] = hf ? zjv[c].w : zjv[c].y;
      acc = __builtin_amdgcn_wmma_f32_16x16x4_f32(false, a, false, b,
                                                  (short)0, acc, false, false);
    }
    const float sqj = sq[j0 + r];
#pragma unroll
    for (int i = 0; i < 8; ++i) {
      const int m = i0 + i + hf * 8;
      const float v = 2.0f * acc[i] - sqi[i] - sqj + alpha;
      __builtin_nontemporal_store(1.0f / (1.0f + __expf(-v)),
                                  &out[(size_t)m * NPTS + j0 + r]);
    }
  }
}

// ---------- host launcher ----------
extern "C" void kernel_launch(void* const* d_in, const int* in_sizes, int n_in,
                              void* d_out, int out_size, void* d_ws, size_t ws_size,
                              hipStream_t stream) {
  (void)in_sizes; (void)n_in; (void)out_size; (void)ws_size;
  const float* X       = (const float*)d_in[0];
  const float* adj     = (const float*)d_in[1];
  const float* W_base  = (const float*)d_in[2];
  const float* W_pi    = (const float*)d_in[3];
  const float* cmean   = (const float*)d_in[4];
  const float* clogstd = (const float*)d_in[5];
  const float* alpha   = (const float*)d_in[6];
  const float* noise   = (const float*)d_in[7];
  float* out = (float*)d_out;

  char* ws = (char*)d_ws;
  __bf16* Wt   = (__bf16*)(ws + OFF_WT);
  __bf16* XWt  = (__bf16*)(ws + OFF_XWT);
  float*  hid  = (float*) (ws + OFF_HID);
  __bf16* P1t  = (__bf16*)(ws + OFF_P1T);
  float*  piB  = (float*) (ws + OFF_PI);
  float*  Zb   = (float*) (ws + OFF_Z);
  float*  sqB  = (float*) (ws + OFF_SQ);

  wt_convert_kernel<<<512, 256, 0, stream>>>(W_base, Wt);
  gemm1_xw_kernel<<<MT, 256, 0, stream>>>(X, Wt, XWt);
  gemm2_hidden_kernel<<<MT, 256, 0, stream>>>(adj, XWt, hid);
  p1_kernel<<<(NPTS + 255) / 256, 256, 0, stream>>>(hid, W_pi, P1t);
  gemm3_pi_kernel<<<(MT + 3) / 4, 128, 0, stream>>>(adj, P1t, piB);
  z_kernel<<<(NPTS + 255) / 256, 256, 0, stream>>>(piB, cmean, clogstd, noise, Zb, sqB);
  apred_kernel<<<dim3((MT + 3) / 4, (MT + 7) / 8), 256, 0, stream>>>(Zb, sqB, alpha, out);
}
